// Transformer_33062658244688
// MI455X (gfx1250) — matrix operand order repaired and checked by reading, not compiled, not run on hardware
//
#include <hip/hip_runtime.h>
#include <hip/hip_bf16.h>
#include <math.h>

// ---------------------------------------------------------------------------
// Transformer forward for MI455X (gfx1250), wave32 + WMMA f32_16x16x32_f16.
// Workspace requirement: ~10 * 25.2MB + small  (~253 MB of d_ws).
// ---------------------------------------------------------------------------

typedef __attribute__((ext_vector_type(16))) _Float16 v16h;
typedef __attribute__((ext_vector_type(8)))  float    v8f;

union F16x16 { v16h v; unsigned u[8]; };
union F8x32  { v8f  v; float    f[8]; };

#define NB    32
#define LL    384
#define WID   512
#define NH    8
#define DH    64
#define C2    1024
#define GG    64      // WIDTH / dimgroup(8)
#define MAXE  384
#define TOTE  (NB*WID*LL)   // 6,291,456 elements in an [N,512,384] tensor

__device__ __forceinline__ unsigned pk2(float a, float b) {
    union { _Float16 h[2]; unsigned u; } t;
    t.h[0] = (_Float16)a; t.h[1] = (_Float16)b;
    return t.u;
}

__device__ __forceinline__ v8f wmma_f16(const F16x16& a, const F16x16& b, v8f c) {
    // 8 args: (neg_a, A, neg_b, B, c_mod, C, reuse_a, reuse_b)
    return __builtin_amdgcn_wmma_f32_16x16x32_f16(false, a.v, false, b.v,
                                                  (short)0, c, false, false);
}

// ---------------------------------------------------------------------------
// Generic batched WMMA GEMM:  Y[b][m][l] = act( sum_k A[m][k] * X[b][k][l] + bias[m] )
// A optionally batched with period a_bmod (stride a_bsize, bias stride bias_bs).
// Block = 128 threads (4 waves). Macrotile 64(M) x 64(L); each wave: 2x2 WMMA tiles.
// Double-buffered LDS: next K-tile global loads issue before the WMMA group,
// convert+ds_store after it, so load latency hides behind the matrix pipes.
// act: 0 = none, 1 = exact GELU.
// ---------------------------------------------------------------------------
__global__ __launch_bounds__(128)
void gemm_wmma_kernel(const float* __restrict__ A, const float* __restrict__ X,
                      float* __restrict__ Y, const float* __restrict__ bias,
                      int M, int K, int Lc,
                      int a_bmod, int a_bsize, int bias_bs,
                      long long x_bstride, long long y_bstride, int act)
{
    __shared__ unsigned AsU[2][64][18];   // 64 rows(M) x 16 u32 (32 f16 K-slice), +pad
    __shared__ unsigned BsU[2][64][18];   // 64 cols(L) x 16 u32, K-transposed

    const int b  = blockIdx.z;
    const int m0 = blockIdx.y * 64;
    const int l0 = blockIdx.x * 64;
    const int ab = (a_bmod > 1) ? (b % a_bmod) : 0;
    const float* Ab = A + (long long)ab * a_bsize;
    const float* Xb = X + (long long)b * x_bstride;
    float*       Yb = Y + (long long)b * y_bstride;

    const int tid  = threadIdx.x;
    const int lane = tid & 31;
    const int wv   = tid >> 5;
    const int wm   = (wv >> 1) * 32;    // wave row offset in macrotile
    const int wn   = (wv & 1) * 32;     // wave col offset in macrotile
    const int hl   = lane >> 4;         // lane half
    const int ln   = lane & 15;

    const v8f vz = {0.f,0.f,0.f,0.f,0.f,0.f,0.f,0.f};
    v8f acc[2][2];
    acc[0][0]=vz; acc[0][1]=vz; acc[1][0]=vz; acc[1][1]=vz;

    const int nk = K >> 5;

    // Prologue: stage K-tile 0 into buffer 0.
    #pragma unroll
    for (int it = 0; it < 8; ++it) {
        int idx = it * 128 + tid;             // 0..1023
        int r = idx >> 4, c = idx & 15;       // row, u32 col (K pair)
        const float* ap = Ab + (long long)(m0 + r) * K + c * 2;
        AsU[0][r][c] = pk2(ap[0], ap[1]);
    }
    #pragma unroll
    for (int it = 0; it < 8; ++it) {
        int idx = it * 128 + tid;
        int nn = idx & 63, c = idx >> 6;      // col, K pair
        const float* xp0 = Xb + (long long)(c * 2) * Lc + l0 + nn;
        BsU[0][nn][c] = pk2(xp0[0], xp0[Lc]);
    }

    float pfa[16], pfb[16];
    for (int kt = 0; kt < nk; ++kt) {
        const int cur = kt & 1;
        __syncthreads();

        const bool have_next = (kt + 1 < nk);
        const int k1 = (kt + 1) << 5;
        if (have_next) {
            // Issue next tile's global loads now; consume after the WMMA group.
            #pragma unroll
            for (int it = 0; it < 8; ++it) {
                int idx = it * 128 + tid;
                int r = idx >> 4, c = idx & 15;
                const float* ap = Ab + (long long)(m0 + r) * K + k1 + c * 2;
                pfa[it * 2] = ap[0]; pfa[it * 2 + 1] = ap[1];
            }
            #pragma unroll
            for (int it = 0; it < 8; ++it) {
                int idx = it * 128 + tid;
                int nn = idx & 63, c = idx >> 6;
                const float* xp0 = Xb + (long long)(k1 + c * 2) * Lc + l0 + nn;
                pfb[it * 2] = xp0[0]; pfb[it * 2 + 1] = xp0[Lc];
            }
            if (k1 + 32 < K) {
                // Warm L2 two K-tiles ahead (gfx1250 global_prefetch_b8).
                __builtin_prefetch(Ab + (long long)(m0 + (tid >> 1)) * K + k1 + 32, 0, 1);
                __builtin_prefetch(Xb + (long long)(k1 + 32 + (tid >> 2)) * Lc + l0, 0, 1);
            }
        }

        // Fragments from current buffer (contiguous-k gathers) + 2x2 WMMA.
        F16x16 af[2], bf[2];
        #pragma unroll
        for (int i = 0; i < 2; ++i) {
            int r = wm + i * 16 + ln;
            #pragma unroll
            for (int vv = 0; vv < 8; ++vv) {
                // A layout: k = half*16 + 2v
                int c = hl * 8 + vv;
                af[i].u[vv] = AsU[cur][r][c];
            }
        }
        #pragma unroll
        for (int j = 0; j < 2; ++j) {
            int nn = wn + j * 16 + ln;
            #pragma unroll
            for (int vv = 0; vv < 8; ++vv)        // B layout: k = half*16 + 2v
                bf[j].u[vv] = BsU[cur][nn][hl * 8 + vv];
        }
        #pragma unroll
        for (int i = 0; i < 2; ++i)
            #pragma unroll
            for (int j = 0; j < 2; ++j)
                acc[i][j] = __builtin_amdgcn_wmma_f32_16x16x32_f16(
                    false, af[i].v, false, bf[j].v, (short)0, acc[i][j], false, false);

        if (have_next) {
            const int nxt = cur ^ 1;
            #pragma unroll
            for (int it = 0; it < 8; ++it) {
                int idx = it * 128 + tid;
                AsU[nxt][idx >> 4][idx & 15] = pk2(pfa[it * 2], pfa[it * 2 + 1]);
            }
            #pragma unroll
            for (int it = 0; it < 8; ++it) {
                int idx = it * 128 + tid;
                BsU[nxt][idx & 63][idx >> 6] = pk2(pfb[it * 2], pfb[it * 2 + 1]);
            }
        }
    }

    const int biasOff = (a_bmod > 1) ? (ab * bias_bs) : 0;
    #pragma unroll
    for (int i = 0; i < 2; ++i)
        #pragma unroll
        for (int j = 0; j < 2; ++j) {
            F8x32 t; t.v = acc[i][j];
            int colg = l0 + wn + j * 16 + ln;
            #pragma unroll
            for (int r = 0; r < 8; ++r) {
                int rowm = m0 + wm + i * 16 + r + hl * 8;
                float y = t.f[r];
                if (bias) y += bias[biasOff + rowm];
                if (act == 1) y = 0.5f * y * (1.0f + erff(y * 0.70710678118654752f));
                Yb[(long long)rowm * Lc + colg] = y;
            }
        }
}

// ---------------------------------------------------------------------------
// Attention: one wave per (n, h, 16-query tile). Flash-style: scores [16,384]
// in LDS via WMMA, fused bias/gate/mask/norm, softmax, then P @ V^T via WMMA.
// ---------------------------------------------------------------------------
__global__ __launch_bounds__(32)
void attn_kernel(const float* __restrict__ qm, const float* __restrict__ km,
                 const float* __restrict__ vm, const float* __restrict__ relpos,
                 const float* __restrict__ gatelog, const float* __restrict__ maskb,
                 const float* __restrict__ normb, float* __restrict__ obuf)
{
    __shared__ float S[16][392];   // padded rows

    const int n = blockIdx.z, h = blockIdx.y, q0 = blockIdx.x * 16;
    const int lane = threadIdx.x;
    const int ln = lane & 15, hl = lane >> 4;
    const long long headoff = ((long long)n * WID + h * DH) * LL;
    const v8f vz = {0.f,0.f,0.f,0.f,0.f,0.f,0.f,0.f};

    // A fragments: A[m=query, k=d] = q[d, q0+m]   (D=64 -> two 16x32 frags)
    F16x16 aq[2];
    {
        const float* qb = qm + headoff + q0 + ln;
        #pragma unroll
        for (int c = 0; c < 2; ++c)
            #pragma unroll
            for (int vv = 0; vv < 8; ++vv) {
                int k2 = hl * 16 + 2 * vv;
                int kk = c * 32 + k2;
                aq[c].u[vv] = pk2(qb[(long long)kk * LL], qb[(long long)(kk + 1) * LL]);
            }
    }
    const float nrm = normb[n];

    // ---- scores ----
    for (int j0 = 0; j0 < LL; j0 += 16) {
        F16x16 bk[2];
        const float* kb = km + headoff + j0 + ln;
        #pragma unroll
        for (int c = 0; c < 2; ++c)
            #pragma unroll
            for (int vv = 0; vv < 8; ++vv) {
                int kk = c * 32 + hl * 16 + 2 * vv;
                bk[c].u[vv] = pk2(kb[(long long)kk * LL], kb[(long long)(kk + 1) * LL]);
            }
        v8f acc = vz;
        acc = wmma_f16(aq[0], bk[0], acc);
        acc = wmma_f16(aq[1], bk[1], acc);

        F8x32 t; t.v = acc;
        int jg = j0 + ln;
        float gv = gatelog[((long long)n * NH + h) * LL + jg];
        float mv = maskb[n * LL + jg];
        #pragma unroll
        for (int r = 0; r < 8; ++r) {
            int ig = q0 + r + hl * 8;
            int idx = MAXE - ig + jg;
            idx = idx < 0 ? 0 : (idx > 2 * MAXE - 2 ? 2 * MAXE - 2 : idx);
            S[r + hl * 8][jg] = (t.f[r] + relpos[idx] + gv + mv) / nrm;
        }
    }
    __syncthreads();

    // ---- softmax over keys, rows owned by lanes 0..15 ----
    if (lane < 16) {
        float mx = -1e30f;
        for (int j = 0; j < LL; ++j) mx = fmaxf(mx, S[lane][j]);
        float sm = 0.f;
        for (int j = 0; j < LL; ++j) { float e = __expf(S[lane][j] - mx); S[lane][j] = e; sm += e; }
        float inv = 1.f / sm;
        for (int j = 0; j < LL; ++j) S[lane][j] *= inv;
    }
    __syncthreads();

    // ---- O = P @ V^T : M=16 queries, N=64 dims, K=384 keys ----
    v8f oacc[4] = {vz, vz, vz, vz};
    for (int k0 = 0; k0 < LL; k0 += 32) {
        F16x16 ap;
        #pragma unroll
        for (int vv = 0; vv < 8; ++vv) {
            int k2 = hl * 16 + 2 * vv;
            ap.u[vv] = pk2(S[ln][k0 + k2], S[ln][k0 + k2 + 1]);
        }
        #pragma unroll
        for (int t4 = 0; t4 < 4; ++t4) {
            F16x16 bv;
            const float* vb = vm + headoff + (long long)(t4 * 16 + ln) * LL + k0;
            #pragma unroll
            for (int vv = 0; vv < 8; ++vv) {
                int kk = hl * 16 + 2 * vv;
                bv.u[vv] = pk2(vb[kk], vb[kk + 1]);
            }
            oacc[t4] = wmma_f16(ap, bv, oacc[t4]);
        }
    }
    #pragma unroll
    for (int t4 = 0; t4 < 4; ++t4) {
        F8x32 t; t.v = oacc[t4];
        int d = t4 * 16 + ln;
        #pragma unroll
        for (int r = 0; r < 8; ++r)
            obuf[headoff + (long long)d * LL + q0 + r + hl * 8] = t.f[r];
    }
}

// ---------------------------------------------------------------------------
// Elementwise / small kernels
// ---------------------------------------------------------------------------
__global__ void embed_kernel(const int* __restrict__ xtok, const int* __restrict__ pos,
                             const float* __restrict__ embed, const float* __restrict__ pe0,
                             const float* __restrict__ pe1,
                             float* __restrict__ xe, float* __restrict__ xp)
{
    int i = blockIdx.x * blockDim.x + threadIdx.x;
    if (i >= TOTE) return;
    int l = i % LL, c = (i / LL) % WID, n = i / (WID * LL);
    int tok = xtok[n * LL + l];
    xe[i] = embed[tok * WID + c];
    if (c < 256) {
        int p = pos[n * 3 + 0] + l; if (p > MAXE) p = MAXE;
        xp[i] = pe0[p * 256 + c];
    } else {
        int p = pos[n * 3 + 1] + l; if (p > MAXE) p = MAXE;
        xp[i] = pe1[p * 256 + (c - 256)];
    }
}

__global__ void masknorm_kernel(const int* __restrict__ pos,
                                float* __restrict__ maskb, float* __restrict__ normb)
{
    int i = blockIdx.x * blockDim.x + threadIdx.x;
    if (i >= NB * LL) return;
    int n = i / LL, l = i % LL;
    int vl = pos[n * 3 + 2];
    maskb[i] = (l >= vl) ? -1e30f : 0.f;
    if (l == 0) normb[n] = sqrtf((float)vl);
}

__global__ void prep_qkv_kernel(const float* __restrict__ x, const float* __restrict__ xe,
                                const float* __restrict__ xp,
                                const float* __restrict__ qkorg, const float* __restrict__ qkpos,
                                const float* __restrict__ vorg,
                                float* __restrict__ x1, float* __restrict__ x0)
{
    int i = blockIdx.x * blockDim.x + threadIdx.x;
    if (i >= TOTE) return;
    int g = ((i / LL) % WID) >> 3;
    float xv = x[i], ev = xe[i];
    x1[i] = xv + ev * qkorg[g] + xp[i] * qkpos[g];
    x0[i] = xv + ev * vorg[g];
}

__global__ void prep_dense_kernel(const float* __restrict__ x, const float* __restrict__ xe,
                                  const float* __restrict__ xorg, float* __restrict__ t1)
{
    int i = blockIdx.x * blockDim.x + threadIdx.x;
    if (i >= TOTE) return;
    int g = ((i / LL) % WID) >> 3;
    t1[i] = x[i] + xe[i] * xorg[g];
}

__global__ void gate8_kernel(const float* __restrict__ gw, const float* __restrict__ gb,
                             const float* __restrict__ x, float* __restrict__ g)
{
    int i = blockIdx.x * blockDim.x + threadIdx.x;
    if (i >= NB * NH * LL) return;
    int l = i % LL, h = (i / LL) % NH, n = i / (NH * LL);
    const float* xr = x + (long long)n * WID * LL + l;
    const float* wr = gw + h * WID;
    float s = gb[h];
    for (int w = 0; w < WID; ++w) s += wr[w] * xr[(long long)w * LL];
    g[i] = s;
}

// y = x + v*sigmoid(glog+gb)*res ; x <- GroupNorm(1,C)(y). One block per sample.
__global__ __launch_bounds__(256)
void gate_norm_kernel(float* __restrict__ x, const float* __restrict__ vval,
                      const float* __restrict__ glog,
                      const float* __restrict__ res, const float* __restrict__ gb)
{
    __shared__ float r1[256], r2[256];
    const int n = blockIdx.x, tid = threadIdx.x;
    const int CL = WID * LL;
    float*       xb = x    + (long long)n * CL;
    const float* vb = vval + (long long)n * CL;
    const float* gl = glog + (long long)n * GG * LL;

    float s = 0.f, s2 = 0.f;
    for (int i = tid; i < CL; i += 256) {
        int g = (i / LL) >> 3, l = i % LL;
        float gg = 1.f / (1.f + __expf(-(gl[g * LL + l] + gb[g])));
        float y = xb[i] + vb[i] * gg * res[g];
        s += y; s2 += y * y;
    }
    r1[tid] = s; r2[tid] = s2;
    __syncthreads();
    for (int st = 128; st > 0; st >>= 1) {
        if (tid < st) { r1[tid] += r1[tid + st]; r2[tid] += r2[tid + st]; }
        __syncthreads();
    }
    float mean = r1[0] / CL;
    float var  = r2[0] / CL - mean * mean;
    float inv  = rsqrtf(var + 1e-5f);
    for (int i = tid; i < CL; i += 256) {
        int g = (i / LL) >> 3, l = i % LL;
        float gg = 1.f / (1.f + __expf(-(gl[g * LL + l] + gb[g])));
        float y = xb[i] + vb[i] * gg * res[g];
        xb[i] = (y - mean) * inv;
    }
}

__global__ void head_kernel(const float* __restrict__ x, const float* __restrict__ hw,
                            const float* __restrict__ hb, float* __restrict__ out)
{
    int i = blockIdx.x * blockDim.x + threadIdx.x;
    if (i >= NB * LL * 28) return;
    int o = i % 28, l = (i / 28) % LL, n = i / (28 * LL);
    const float* xr = x + (long long)n * WID * LL + l;
    const float* wr = hw + o * WID;
    float s = hb[o];
    for (int w = 0; w < WID; ++w) s += xr[(long long)w * LL] * wr[w];
    out[i] = s;
}

// ---------------------------------------------------------------------------
extern "C" void kernel_launch(void* const* d_in, const int* in_sizes, int n_in,
                              void* d_out, int out_size, void* d_ws, size_t ws_size,
                              hipStream_t stream)
{
    (void)in_sizes; (void)n_in; (void)out_size; (void)ws_size;

    const int*   xtok  = (const int*)d_in[0];
    const int*   pos   = (const int*)d_in[1];
    const float* embed = (const float*)d_in[2];
    const float* pe0   = (const float*)d_in[3];
    const float* pe1   = (const float*)d_in[4];
    auto LP = [&](int layer, int idx) -> const float* {
        return (const float*)d_in[5 + layer * 26 + idx];
    };

    float* ws = (float*)d_ws;
    const size_t SZ = (size_t)TOTE;
    float* xe    = ws;            // [N,512,384]
    float* xp    = xe + SZ;
    float* x     = xp + SZ;
    float* t1    = x + SZ;
    float* t0    = t1 + SZ;
    float* q     = t0 + SZ;       // q..k doubles as big1 [N,1024,384] in dense stage
    float* k     = q + SZ;
    float* v     = k + SZ;
    float* big2  = v + SZ;        // [N,1024,384]
    float* maskb = big2 + 2 * SZ;
    float* normb = maskb + NB * LL;
    float* g8    = normb + NB;            // [N,8,384]
    float* glog  = g8 + NB * NH * LL;     // [N,64,384]

    float* xsOut = (float*)d_out;                     // [4,N,512,384]
    float* ysOut = xsOut + (size_t)4 * SZ;            // [4,N,384,28]

    const long long SB = (long long)WID * LL;         // per-sample stride 512*384

    embed_kernel<<<(TOTE + 255) / 256, 256, 0, stream>>>(xtok, pos, embed, pe0, pe1, xe, xp);
    masknorm_kernel<<<(NB * LL + 255) / 256, 256, 0, stream>>>(pos, maskb, normb);
    hipMemcpyAsync(x, xe, SZ * sizeof(float), hipMemcpyDeviceToDevice, stream);

    for (int layer = 0; layer < 4; ++layer) {
        const float* qkpos0 = LP(layer, 0);
        const float* qkorg0 = LP(layer, 1);
        const float* vorg0  = LP(layer, 2);
        const float* relpos = LP(layer, 3);
        const float* gate_w = LP(layer, 4);
        const float* gate_b = LP(layer, 5);
        const float* q_w    = LP(layer, 6);
        const float* k_w    = LP(layer, 7);
        const float* v_w    = LP(layer, 8);
        const float* dw     = LP(layer, 9);
        const float* db     = LP(layer, 10);
        const float* a0_res = LP(layer, 11);
        const float* a0_gw  = LP(layer, 12);
        const float* a0_gb  = LP(layer, 13);
        const float* xorg0  = LP(layer, 14);
        const float* f1_w   = LP(layer, 15);
        const float* f1_b   = LP(layer, 16);
        const float* f2_w   = LP(layer, 17);
        const float* f2_b   = LP(layer, 18);
        const float* f3_w   = LP(layer, 19);
        const float* f3_b   = LP(layer, 20);
        const float* d0_res = LP(layer, 21);
        const float* d0_gw  = LP(layer, 22);
        const float* d0_gb  = LP(layer, 23);
        const float* h_w    = LP(layer, 24);
        const float* h_b    = LP(layer, 25);

        // ---- attention stage ----
        prep_qkv_kernel<<<(TOTE + 255) / 256, 256, 0, stream>>>(x, xe, xp, qkorg0, qkpos0, vorg0, t1, t0);
        dim3 g512(LL / 64, WID / 64, NB);
        gemm_wmma_kernel<<<g512, 128, 0, stream>>>(q_w, t1, q, nullptr, 512, 512, LL, 1, 0, 0, SB, SB, 0);
        gemm_wmma_kernel<<<g512, 128, 0, stream>>>(k_w, t1, k, nullptr, 512, 512, LL, 1, 0, 0, SB, SB, 0);
        gemm_wmma_kernel<<<g512, 128, 0, stream>>>(v_w, t0, v, nullptr, 512, 512, LL, 1, 0, 0, SB, SB, 0);
        gate8_kernel<<<(NB * NH * LL + 255) / 256, 256, 0, stream>>>(gate_w, gate_b, x, g8);
        attn_kernel<<<dim3(LL / 16, NH, NB), 32, 0, stream>>>(q, k, v, relpos, g8, maskb, normb, t1);
        gemm_wmma_kernel<<<g512, 128, 0, stream>>>(dw, t1, t0, db, 512, 512, LL, 1, 0, 0, SB, SB, 0);
        gemm_wmma_kernel<<<dim3(LL / 64, 1, NB), 128, 0, stream>>>(a0_gw, x, glog, nullptr,
                                                                   64, 512, LL, 1, 0, 0, SB, (long long)GG * LL, 0);
        gate_norm_kernel<<<NB, 256, 0, stream>>>(x, t0, glog, a0_res, a0_gb);

        // ---- dense stage ----
        prep_dense_kernel<<<(TOTE + 255) / 256, 256, 0, stream>>>(x, xe, xorg0, t1);
        gemm_wmma_kernel<<<dim3(LL / 64, C2 / 64, NB), 128, 0, stream>>>(f1_w, t1, q /*big1*/, f1_b,
                                                                         C2, 512, LL, 1, 0, 0, SB, (long long)C2 * LL, 1);
        gemm_wmma_kernel<<<dim3(LL / 64, 128 / 64, NB * NH), 128, 0, stream>>>(f2_w, q, big2, f2_b,
                                                                               128, 128, LL, NH, 128 * 128, 128,
                                                                               (long long)128 * LL, (long long)128 * LL, 1);
        gemm_wmma_kernel<<<dim3(LL / 64, WID / 64, NB), 128, 0, stream>>>(f3_w, big2, t0, f3_b,
                                                                          512, C2, LL, 1, 0, 0, (long long)C2 * LL, SB, 0);
        gemm_wmma_kernel<<<dim3(LL / 64, 1, NB), 128, 0, stream>>>(d0_gw, x, glog, nullptr,
                                                                   64, 512, LL, 1, 0, 0, SB, (long long)GG * LL, 0);
        gate_norm_kernel<<<NB, 256, 0, stream>>>(x, t0, glog, d0_res, d0_gb);

        // ---- per-layer outputs ----
        head_kernel<<<(NB * LL * 28 + 255) / 256, 256, 0, stream>>>(x, h_w, h_b,
                                                                    ysOut + (size_t)layer * NB * LL * 28);
        hipMemcpyAsync(xsOut + (size_t)layer * SZ, x, SZ * sizeof(float),
                       hipMemcpyDeviceToDevice, stream);
    }
}
